// QKNormAttention_65472481460332
// MI455X (gfx1250) — matrix operand order repaired and checked by reading, not compiled, not run on hardware
//
#include <hip/hip_runtime.h>
#include <hip/hip_bf16.h>

// ---------------------------------------------------------------------------
// QK-Norm attention block for MI455X (gfx1250, wave32, WMMA).
// B=4, N=2048, D=1024, H=16, hd=64.
// All GEMM-like math on v_wmma_f32_16x16x32_f16; softmax/LN in f32 VALU.
// Whole problem (~90MB) is L2-resident (192MB), so kernels stream operands
// straight from global/L2 into WMMA fragments; only the softmax P-matrix
// C->A fragment relayout goes through LDS.
// ---------------------------------------------------------------------------

typedef __attribute__((ext_vector_type(16))) _Float16 v16h;
typedef __attribute__((ext_vector_type(8)))  _Float16 v8h;
typedef __attribute__((ext_vector_type(8)))  float    v8f;

union V16U { v16h v; v8h h[2]; };
union H2U  { _Float16 h[2]; unsigned int u; };

#define BATCH   4
#define SEQ     2048
#define DIM     1024
#define NHEAD   16
#define HDIM    64
#define MROWS   (BATCH * SEQ)          // 8192
#define QKVDIM  (3 * DIM)              // 3072

// -------- WMMA helpers (CDNA5 fragment layouts, 16x16x32 f16) ---------------

__device__ __forceinline__ v8f wmma_f16(v16h a, v16h b, v8f c) {
  // D = A(16x32) * B(32x16) + C(16x16 f32)
  return __builtin_amdgcn_wmma_f32_16x16x32_f16(
      false, a, false, b, (short)0, c, false, false);
}

// A fragment: 16x32 f16 tile, row-major with leading dim ld (elements).
// lane 0-15: row=lane,   halfs[0..15]=K 0..15
// lane16-31: row=lane-16,halfs[0..15]=K 16..31
__device__ __forceinline__ v16h load_frag_a(const _Float16* __restrict__ base,
                                            int ld, int lane) {
  int row  = lane & 15;
  int koff = (lane < 16) ? 0 : 16;
  const _Float16* p = base + (size_t)row * ld + koff;
  V16U u;
  u.h[0] = *(const v8h*)(p);
  u.h[1] = *(const v8h*)(p + 8);
  return u.v;
}

// B fragment: 32x16 (KxN). Source stored N-major with K contiguous per row
// (i.e. Bt[n][k], leading dim ld). lane col = lane&15,
// K values = (lane<16 ? 0 : 16) + 0..15 contiguous.
__device__ __forceinline__ v16h load_frag_b(const _Float16* __restrict__ baseT,
                                            int ld, int lane) {
  int col  = lane & 15;
  int koff = (lane < 16) ? 0 : 16;
  return *(const v16h*)(baseT + (size_t)col * ld + koff);
}

// -------- Kernel 1: LayerNorm, f32 in -> f16 out ----------------------------

__global__ void __launch_bounds__(256)
ln_kernel(const float* __restrict__ x, const float* __restrict__ w,
          const float* __restrict__ b, _Float16* __restrict__ h) {
  int row = blockIdx.x;                       // 8192 rows
  const float* xr = x + (size_t)row * DIM;
  __shared__ float redS[8], redS2[8];
  float vals[4];
  float s = 0.f, s2 = 0.f;
#pragma unroll
  for (int i = 0; i < 4; ++i) {
    float v = xr[threadIdx.x + i * 256];
    vals[i] = v; s += v; s2 += v * v;
  }
#pragma unroll
  for (int m = 16; m >= 1; m >>= 1) {
    s  += __shfl_xor(s,  m, 32);
    s2 += __shfl_xor(s2, m, 32);
  }
  int wv = threadIdx.x >> 5;
  if ((threadIdx.x & 31) == 0) { redS[wv] = s; redS2[wv] = s2; }
  __syncthreads();
  s = 0.f; s2 = 0.f;
#pragma unroll
  for (int i = 0; i < 8; ++i) { s += redS[i]; s2 += redS2[i]; }
  float mu   = s * (1.0f / DIM);
  float var  = s2 * (1.0f / DIM) - mu * mu;
  float rstd = rsqrtf(var + 1e-5f);
#pragma unroll
  for (int i = 0; i < 4; ++i) {
    int c = threadIdx.x + i * 256;
    float y = (vals[i] - mu) * rstd * w[c] + b[c];
    h[(size_t)row * DIM + c] = (_Float16)y;
  }
}

// -------- Kernel 2: f32 -> f16 convert --------------------------------------

__global__ void __launch_bounds__(256)
cvt_f32_f16(const float* __restrict__ in, _Float16* __restrict__ out, int n) {
  for (int i = blockIdx.x * 256 + threadIdx.x; i < n; i += gridDim.x * 256)
    out[i] = (_Float16)in[i];
}

// -------- Kernel 3: QKV GEMM (h @ Wqkv^T + b), scatter to q/k/vT ------------
// grid (MROWS/32, QKVDIM/512), block 256 (8 waves). Wave tile: 32(M) x 64(N).
// Epilogue is branch-free: a 16-wide C fragment never straddles a q/k/v
// 1024-column boundary and a 32-row block never straddles a batch boundary,
// so destination pointer/offset/stride are per-fragment selects.

__global__ void __launch_bounds__(256)
qkv_gemm(const _Float16* __restrict__ hbuf, const _Float16* __restrict__ w16,
         const float* __restrict__ bias, _Float16* __restrict__ qbuf,
         _Float16* __restrict__ kbuf, _Float16* __restrict__ vT) {
  int lane = threadIdx.x & 31, wv = threadIdx.x >> 5;
  int m0 = blockIdx.x * 32;
  int n0 = blockIdx.y * 512 + wv * 64;

  v8f acc[2][4] = {};
  const _Float16* Abase = hbuf + (size_t)m0 * DIM;

  for (int k0 = 0; k0 < DIM; k0 += 32) {
    __builtin_prefetch((const void*)(Abase + k0 + 128), 0, 1);
    v16h a[2], bfr[4];
#pragma unroll
    for (int mt = 0; mt < 2; ++mt)
      a[mt] = load_frag_a(Abase + (size_t)(mt * 16) * DIM + k0, DIM, lane);
#pragma unroll
    for (int nt = 0; nt < 4; ++nt)
      bfr[nt] = load_frag_b(w16 + (size_t)(n0 + nt * 16) * DIM + k0, DIM, lane);
#pragma unroll
    for (int mt = 0; mt < 2; ++mt)
#pragma unroll
      for (int nt = 0; nt < 4; ++nt)
        acc[mt][nt] = wmma_f16(a[mt], bfr[nt], acc[mt][nt]);
  }

  int rowadd = (lane < 16) ? 0 : 8;
  int bb     = m0 >> 11;                      // batch index (block-uniform)
  int nseq0  = m0 & 2047;                     // sequence row base
#pragma unroll
  for (int nt = 0; nt < 4; ++nt) {
    int c   = n0 + nt * 16 + (lane & 15);     // absolute output column
    int i   = c >> 10;                        // 0=q 1=k 2=v (fragment-wide)
    int rem = c & 1023;
    int hh  = rem >> 6;
    int d   = rem & 63;
    int bh  = bb * NHEAD + hh;
    float bv = bias[c];
    _Float16* outp = (i == 0) ? qbuf : ((i == 1) ? kbuf : vT);
    size_t base = (i < 2) ? ((size_t)bh * (SEQ * HDIM) + (size_t)d)
                          : (((size_t)bh * HDIM + d) * SEQ);
    int stepn = (i < 2) ? HDIM : 1;           // stride per sequence row
#pragma unroll
    for (int mt = 0; mt < 2; ++mt) {
      int nb = nseq0 + mt * 16 + rowadd;
#pragma unroll
      for (int r = 0; r < 8; ++r) {
        _Float16 val = (_Float16)(acc[mt][nt][r] + bv);
        outp[base + (size_t)(nb + r) * stepn] = val;
      }
    }
  }
}

// -------- Kernel 4: per-head L2 normalize (q and k, contiguous) -------------
// One wave per 64-element row; rows = 2*B*H*N. Packed b32 access per lane.

__global__ void __launch_bounds__(256)
l2norm_kernel(_Float16* __restrict__ buf) {
  int row  = blockIdx.x * 8 + (threadIdx.x >> 5);
  int lane = threadIdx.x & 31;
  unsigned int* p32 = (unsigned int*)(buf + ((size_t)row << 6)) + lane;
  H2U u; u.u = *p32;
  float a = (float)u.h[0], b = (float)u.h[1];
  float s = a * a + b * b;
#pragma unroll
  for (int m = 16; m >= 1; m >>= 1) s += __shfl_xor(s, m, 32);
  float inv = 1.0f / fmaxf(sqrtf(s), 1e-12f);
  u.h[0] = (_Float16)(a * inv);
  u.h[1] = (_Float16)(b * inv);
  *p32 = u.u;
}

// -------- Kernel 5: flash attention per (b,h) -------------------------------
// grid = B*H*(SEQ/64), block 128 (4 waves), wave owns one 16-row Q tile.

__global__ void __launch_bounds__(128)
attn_kernel(const _Float16* __restrict__ qbuf, const _Float16* __restrict__ kbuf,
            const _Float16* __restrict__ vT, _Float16* __restrict__ abuf) {
  __shared__ _Float16 plds[4][16 * 32];
  int lane = threadIdx.x & 31, wv = threadIdx.x >> 5;
  int qblk = blockIdx.x & 31;
  int bh   = blockIdx.x >> 5;
  int q0   = qblk * 64 + wv * 16;

  const _Float16* qb  = qbuf + (size_t)bh * SEQ * HDIM;
  const _Float16* kb_ = kbuf + (size_t)bh * SEQ * HDIM;
  const _Float16* vb  = vT   + (size_t)bh * HDIM * SEQ;
  _Float16* pl = plds[wv];

  v16h qa0 = load_frag_a(qb + (size_t)q0 * HDIM + 0,  HDIM, lane);
  v16h qa1 = load_frag_a(qb + (size_t)q0 * HDIM + 32, HDIM, lane);

  v8f o[4] = {};
  float mrow[8], lrow[8];
#pragma unroll
  for (int r = 0; r < 8; ++r) { mrow[r] = -1e30f; lrow[r] = 0.f; }

  int colbase = lane & 15;
  int rowadd  = (lane < 16) ? 0 : 8;

  for (int kb0 = 0; kb0 < SEQ; kb0 += 32) {
    __builtin_prefetch((const void*)(kb_ + (size_t)(kb0 + 32) * HDIM), 0, 1);
    // ---- S = Q * K^T (scaled) --------------------------------------------
    v8f s[2] = {};
#pragma unroll
    for (int t = 0; t < 2; ++t) {
      v16h b0 = load_frag_b(kb_ + (size_t)(kb0 + t * 16) * HDIM + 0,  HDIM, lane);
      v16h b1 = load_frag_b(kb_ + (size_t)(kb0 + t * 16) * HDIM + 32, HDIM, lane);
      s[t] = wmma_f16(qa0, b0, s[t]);
      s[t] = wmma_f16(qa1, b1, s[t]);
    }
#pragma unroll
    for (int t = 0; t < 2; ++t)
#pragma unroll
      for (int r = 0; r < 8; ++r) s[t][r] *= 8.0f;   // * sqrt(hd)

    // ---- online softmax ---------------------------------------------------
    float p0[8], p1[8];
#pragma unroll
    for (int r = 0; r < 8; ++r) {
      float v = fmaxf(s[0][r], s[1][r]);
#pragma unroll
      for (int m = 8; m >= 1; m >>= 1) v = fmaxf(v, __shfl_xor(v, m, 16));
      float newm = fmaxf(mrow[r], v);
      p0[r] = __expf(s[0][r] - newm);
      p1[r] = __expf(s[1][r] - newm);
      float rs = p0[r] + p1[r];
#pragma unroll
      for (int m = 8; m >= 1; m >>= 1) rs += __shfl_xor(rs, m, 16);
      float alpha = __expf(mrow[r] - newm);
      lrow[r] = lrow[r] * alpha + rs;
      mrow[r] = newm;
#pragma unroll
      for (int f = 0; f < 4; ++f) o[f][r] *= alpha;
    }

    // ---- P: C-fragment -> LDS -> A-fragment ------------------------------
#pragma unroll
    for (int r = 0; r < 8; ++r) {
      pl[(r + rowadd) * 32 + colbase]      = (_Float16)p0[r];
      pl[(r + rowadd) * 32 + 16 + colbase] = (_Float16)p1[r];
    }
    v16h pa;
    {
      int prow = lane & 15;
      int c1   = (lane < 16) ? 0 : 16;
      V16U u;
      u.h[0] = *(const v8h*)(pl + prow * 32 + c1);
      u.h[1] = *(const v8h*)(pl + prow * 32 + 8 + c1);
      pa = u.v;
    }

    // ---- O += P * V  (V^T layout: contiguous K per column) ---------------
#pragma unroll
    for (int f = 0; f < 4; ++f) {
      v16h bv = load_frag_b(vb + (size_t)(f * 16) * SEQ + kb0, SEQ, lane);
      o[f] = wmma_f16(pa, bv, o[f]);
    }
  }

  // ---- finalize: divide by row sum, store f16 [B,N,D] ---------------------
  int b  = bh >> 4, hh = bh & 15;
#pragma unroll
  for (int f = 0; f < 4; ++f) {
    int d = hh * 64 + f * 16 + colbase;
#pragma unroll
    for (int r = 0; r < 8; ++r) {
      int row = q0 + r + rowadd;
      float val = o[f][r] / lrow[r];
      abuf[((size_t)(b * SEQ + row) << 10) + d] = (_Float16)val;
    }
  }
}

// -------- Kernel 6: out projection + bias + residual (f32 out) --------------
// grid (MROWS/32, DIM/512), block 256. NT stores: y is never re-read.

__global__ void __launch_bounds__(256)
out_gemm(const _Float16* __restrict__ abuf, const _Float16* __restrict__ w16,
         const float* __restrict__ bias, const float* __restrict__ x,
         float* __restrict__ y) {
  int lane = threadIdx.x & 31, wv = threadIdx.x >> 5;
  int m0 = blockIdx.x * 32;
  int n0 = blockIdx.y * 512 + wv * 64;

  v8f acc[2][4] = {};
  const _Float16* Abase = abuf + (size_t)m0 * DIM;

  for (int k0 = 0; k0 < DIM; k0 += 32) {
    __builtin_prefetch((const void*)(Abase + k0 + 128), 0, 1);
    v16h a[2], bfr[4];
#pragma unroll
    for (int mt = 0; mt < 2; ++mt)
      a[mt] = load_frag_a(Abase + (size_t)(mt * 16) * DIM + k0, DIM, lane);
#pragma unroll
    for (int nt = 0; nt < 4; ++nt)
      bfr[nt] = load_frag_b(w16 + (size_t)(n0 + nt * 16) * DIM + k0, DIM, lane);
#pragma unroll
    for (int mt = 0; mt < 2; ++mt)
#pragma unroll
      for (int nt = 0; nt < 4; ++nt)
        acc[mt][nt] = wmma_f16(a[mt], bfr[nt], acc[mt][nt]);
  }

  int rowadd = (lane < 16) ? 0 : 8;
#pragma unroll
  for (int mt = 0; mt < 2; ++mt) {
#pragma unroll
    for (int nt = 0; nt < 4; ++nt) {
      int n = n0 + nt * 16 + (lane & 15);
      float bv = bias[n];
#pragma unroll
      for (int r = 0; r < 8; ++r) {
        int m = m0 + mt * 16 + r + rowadd;
        size_t idx = ((size_t)m << 10) + n;
        __builtin_nontemporal_store(x[idx] + acc[mt][nt][r] + bv, &y[idx]);
      }
    }
  }
}

// ---------------------------------------------------------------------------

extern "C" void kernel_launch(void* const* d_in, const int* in_sizes, int n_in,
                              void* d_out, int out_size, void* d_ws, size_t ws_size,
                              hipStream_t stream) {
  (void)in_sizes; (void)n_in; (void)out_size; (void)ws_size;
  const float* x     = (const float*)d_in[0];
  const float* ln_w  = (const float*)d_in[1];
  const float* ln_b  = (const float*)d_in[2];
  const float* qkv_w = (const float*)d_in[3];
  const float* qkv_b = (const float*)d_in[4];
  const float* out_w = (const float*)d_in[5];
  const float* out_b = (const float*)d_in[6];
  float* y = (float*)d_out;

  // Workspace layout (all 256B-aligned, ~88 MB total)
  char* ws = (char*)d_ws;
  size_t off = 0;
  auto alloc = [&](size_t bytes) {
    char* p = ws + off;
    off += (bytes + 255) & ~(size_t)255;
    return p;
  };
  _Float16* hbuf  = (_Float16*)alloc((size_t)MROWS * DIM * 2);        // 16 MB
  _Float16* wqkv  = (_Float16*)alloc((size_t)QKVDIM * DIM * 2);       //  6 MB
  _Float16* wout  = (_Float16*)alloc((size_t)DIM * DIM * 2);          //  2 MB
  _Float16* qbuf  = (_Float16*)alloc((size_t)MROWS * DIM * 2);        // 16 MB
  _Float16* kbuf  = (_Float16*)alloc((size_t)MROWS * DIM * 2);        // 16 MB (contiguous after qbuf)
  _Float16* vTbuf = (_Float16*)alloc((size_t)MROWS * DIM * 2);        // 16 MB
  _Float16* abuf  = (_Float16*)alloc((size_t)MROWS * DIM * 2);        // 16 MB

  // 1) LayerNorm -> f16 h
  ln_kernel<<<MROWS, 256, 0, stream>>>(x, ln_w, ln_b, hbuf);

  // 2) weight conversion f32 -> f16
  cvt_f32_f16<<<2048, 256, 0, stream>>>(qkv_w, wqkv, QKVDIM * DIM);
  cvt_f32_f16<<<1024, 256, 0, stream>>>(out_w, wout, DIM * DIM);

  // 3) QKV projection + scatter to [b,h,n,d] (q,k) and [b,h,d,n] (v)
  qkv_gemm<<<dim3(MROWS / 32, QKVDIM / 512), 256, 0, stream>>>(
      hbuf, wqkv, qkv_b, qbuf, kbuf, vTbuf);

  // 4) L2 normalize q and k rows (qbuf,kbuf contiguous: 2*B*H*N rows of 64)
  l2norm_kernel<<<(2 * BATCH * NHEAD * SEQ) / 8, 256, 0, stream>>>(qbuf);

  // 5) flash attention per (b,h)
  attn_kernel<<<BATCH * NHEAD * (SEQ / 64), 128, 0, stream>>>(
      qbuf, kbuf, vTbuf, abuf);

  // 6) output projection + bias + residual
  out_gemm<<<dim3(MROWS / 32, DIM / 512), 256, 0, stream>>>(
      abuf, wout, out_b, x, y);
}
